// ggRGCN_43817256354379
// MI455X (gfx1250) — matrix operand order repaired and checked by reading, not compiled, not run on hardware
//
#include <hip/hip_runtime.h>
#include <hip/hip_bf16.h>
#include <cstdint>

// ---------------------------------------------------------------------------
// RGCN (basis decomposition, mean aggr) + p-norm gating, MI455X / gfx1250.
//   Dense math  -> v_wmma_f32_16x16x32_bf16, f32 accumulate, fully unrolled.
//   Edge passes -> b128 gathers + global_atomic_add_f32 (the HBM-bound part).
// Tiling: one wave = 16 output rows x all output cols (8 or 4 16-col tiles);
// A fragment loaded once per K-step, reused across all column tiles.
// ---------------------------------------------------------------------------

typedef __attribute__((ext_vector_type(16))) __bf16 v16bf;
typedef __attribute__((ext_vector_type(8)))  __bf16 v8bf;
typedef __attribute__((ext_vector_type(8)))  float  v8f;

#define DD   128   // hidden dim
#define RR   8     // relations
#define NBAS 4     // bases
#define OUTD 64    // output dim

// ---------------------------------------------------------------------------
// A-fragment builder. 16-bit A 16x32 layout (ISA 7.12.2): lane (m = lane&15,
// half = lane>>4): elems 0..7 -> K = c0..c0+7, elems 8..15 -> K = c0+16..+23,
// where c0 = kBase + 8*half. Two contiguous 32B chunks -> v8f loads; scale s
// folds the 1/max(cnt,1) mean normalization; convertvector -> packed bf16 cvt.
// ---------------------------------------------------------------------------
__device__ __forceinline__ v16bf make_afrag(const float* __restrict__ ap,
                                            int c0, float s) {
  const v8f f0 = *(const v8f*)(ap + c0) * s;
  const v8f f1 = *(const v8f*)(ap + c0 + 16) * s;
  const v8bf lo = __builtin_convertvector(f0, v8bf);
  const v8bf hi = __builtin_convertvector(f1, v8bf);
  return __builtin_shufflevector(lo, hi, 0, 1, 2, 3, 4, 5, 6, 7,
                                 8, 9, 10, 11, 12, 13, 14, 15);
}

// Packed-B fragment: 32B contiguous per (kTile, nTile, lane).
__device__ __forceinline__ v16bf bfrag(const __bf16* __restrict__ Bpk,
                                       int kTile, int NTtot, int nt, int lane) {
  return *(const v16bf*)(Bpk + ((size_t)(kTile * NTtot + nt) * 32 + lane) * 16);
}

__device__ __forceinline__ v8f wmma_bf16(v16bf a, v16bf b, v8f c) {
  return __builtin_amdgcn_wmma_f32_16x16x32_bf16(
      /*neg_a=*/false, a, /*neg_b=*/false, b,
      /*c_mod=*/(short)0, c, /*reuse_a=*/false, /*reuse_b=*/false);
}

// ---------------------------------------------------------------------------
// Layer GEMM: hnew = relu((agg/max(cnt,1)) @ Wcat + h @ root + bias)
// K = 8 relations * 128 + 128; fully unrolled: 288 WMMAs per wave.
// Block = 8 waves, each wave owns one 16-row tile (wave-uniform early-out).
// ---------------------------------------------------------------------------
__global__ void __launch_bounds__(256) ggRGCN_layer_gemm_kernel(
    const float* __restrict__ agg, const float* __restrict__ cnt,
    const __bf16* __restrict__ pkW, const float* __restrict__ h,
    const __bf16* __restrict__ pkRoot, const float* __restrict__ bias,
    float* __restrict__ hnew, int M) {
  const int lane = threadIdx.x;
  const int tile = blockIdx.x * 8 + threadIdx.y;
  if ((tile << 4) >= M) return;              // uniform within the wave
  const int rowBase = tile << 4;
  const int half = lane >> 4;
  const int row = rowBase + (lane & 15);

  v8f acc[8] = {};

  // Per-relation mean scales, hoisted: two b128 loads + 8 v_rcp_f32.
  float sc[RR];
  {
    const float4 cA = *(const float4*)(cnt + (size_t)row * RR);
    const float4 cB = *(const float4*)(cnt + (size_t)row * RR + 4);
    sc[0] = __builtin_amdgcn_rcpf(fmaxf(cA.x, 1.0f));
    sc[1] = __builtin_amdgcn_rcpf(fmaxf(cA.y, 1.0f));
    sc[2] = __builtin_amdgcn_rcpf(fmaxf(cA.z, 1.0f));
    sc[3] = __builtin_amdgcn_rcpf(fmaxf(cA.w, 1.0f));
    sc[4] = __builtin_amdgcn_rcpf(fmaxf(cB.x, 1.0f));
    sc[5] = __builtin_amdgcn_rcpf(fmaxf(cB.y, 1.0f));
    sc[6] = __builtin_amdgcn_rcpf(fmaxf(cB.z, 1.0f));
    sc[7] = __builtin_amdgcn_rcpf(fmaxf(cB.w, 1.0f));
  }

  // Aggregated-neighbor term: 8 relation blocks of 4 kTiles each.
  const float* aggRow = agg + (size_t)row * (RR * DD);
#pragma unroll
  for (int r = 0; r < RR; ++r) {
    const float s = sc[r];
#pragma unroll
    for (int kt = 0; kt < 4; ++kt) {
      const v16bf a = make_afrag(aggRow, r * DD + kt * 32 + (half << 3), s);
#pragma unroll
      for (int nt = 0; nt < 8; ++nt)
        acc[nt] = wmma_bf16(a, bfrag(pkW, r * 4 + kt, 8, nt, lane), acc[nt]);
    }
  }

  // Root (self) term.
  const float* hRow = h + (size_t)row * DD;
#pragma unroll
  for (int kt = 0; kt < 4; ++kt) {
    const v16bf a = make_afrag(hRow, kt * 32 + (half << 3), 1.0f);
#pragma unroll
    for (int nt = 0; nt < 8; ++nt)
      acc[nt] = wmma_bf16(a, bfrag(pkRoot, kt, 8, nt, lane), acc[nt]);
  }

  // Epilogue. C/D layout: elem v -> row rowBase + v + 8*half, col = lane&15.
#pragma unroll
  for (int nt = 0; nt < 8; ++nt) {
    const int col = (nt << 4) + (lane & 15);
    const float bv = bias[col];
#pragma unroll
    for (int v = 0; v < 8; ++v) {
      const int r2 = rowBase + v + (half << 3);
      hnew[(size_t)r2 * DD + col] = fmaxf(acc[nt][v] + bv, 0.0f);
    }
  }
}

// ---------------------------------------------------------------------------
// Dense GEMM: C = relu?(A @ B + bias), A f32 [M x KT*32], B packed bf16,
// Ncols = NT*16. Used for input projection (KT=4,NT=8) and readout (KT=4,NT=4).
// ---------------------------------------------------------------------------
template <int KT, int NT>
__global__ void __launch_bounds__(256) ggRGCN_dense_gemm_kernel(
    const float* __restrict__ A, const __bf16* __restrict__ Bpk,
    const float* __restrict__ bias, float* __restrict__ C, int M, int doRelu) {
  const int lane = threadIdx.x;
  const int tile = blockIdx.x * 8 + threadIdx.y;
  if ((tile << 4) >= M) return;
  const int rowBase = tile << 4;
  const int half = lane >> 4;
  const int row = rowBase + (lane & 15);

  v8f acc[NT] = {};
  const float* aRow = A + (size_t)row * (KT * 32);
#pragma unroll
  for (int kt = 0; kt < KT; ++kt) {
    const v16bf a = make_afrag(aRow, kt * 32 + (half << 3), 1.0f);
#pragma unroll
    for (int nt = 0; nt < NT; ++nt)
      acc[nt] = wmma_bf16(a, bfrag(Bpk, kt, NT, nt, lane), acc[nt]);
  }

#pragma unroll
  for (int nt = 0; nt < NT; ++nt) {
    const int col = (nt << 4) + (lane & 15);
    const float bv = bias[col];
#pragma unroll
    for (int v = 0; v < 8; ++v) {
      const int r2 = rowBase + v + (half << 3);
      float val = acc[nt][v] + bv;
      if (doRelu) val = fmaxf(val, 0.0f);
      C[(size_t)r2 * (NT * 16) + col] = val;
    }
  }
}

// ---------------------------------------------------------------------------
// Pack f32 B [K x Ncols] into fragment-major bf16 (32B chunk per lane):
// lane: col = nTile*16 + (lane&15), elem i -> K = kTile*32 + 8*(lane>>4) + i (+8 if i>=8).
// ---------------------------------------------------------------------------
__global__ void ggRGCN_pack_b_kernel(const float* __restrict__ B, int K,
                                     int Ncols, __bf16* __restrict__ out) {
  const int t = blockIdx.x * blockDim.x + threadIdx.x;
  const int NT = Ncols >> 4;
  const int total = (K >> 5) * NT * 32;
  if (t >= total) return;
  const int lane  = t & 31;
  const int tileI = t >> 5;
  const int kTile = tileI / NT;
  const int nTile = tileI - kTile * NT;
  const int n  = (nTile << 4) + (lane & 15);
  const int kb = (kTile << 5) + ((lane >> 4) << 3);
  __bf16* o = out + (size_t)t * 16;
#pragma unroll
  for (int i = 0; i < 16; ++i)
    o[i] = (__bf16)(B[(size_t)(kb + i + ((i >> 3) << 3)) * Ncols + n]);
}

// Wcat[r*D + d, n] = sum_b comp[r,b] * basis[b,d,n]
__global__ void ggRGCN_relw_kernel(const float* __restrict__ basis,
                                   const float* __restrict__ comp,
                                   float* __restrict__ Wcat) {
  const int idx = blockIdx.x * blockDim.x + threadIdx.x;
  if (idx >= RR * DD * DD) return;
  const int n = idx & (DD - 1);
  const int d = (idx >> 7) & (DD - 1);
  const int r = idx >> 14;
  float acc = 0.0f;
#pragma unroll
  for (int b = 0; b < NBAS; ++b)
    acc += comp[r * NBAS + b] * basis[(size_t)(b * DD + d) * DD + n];
  Wcat[idx] = acc;
}

// One wave per edge: agg[(dst*R+et), :] += h[src, :], cnt[(dst*R+et)] += 1
__global__ void ggRGCN_scatter_agg_kernel(const float* __restrict__ h,
                                          const int* __restrict__ src,
                                          const int* __restrict__ dst,
                                          const int* __restrict__ et,
                                          float* __restrict__ agg,
                                          float* __restrict__ cnt, int E) {
  const int e = blockIdx.x * blockDim.y + threadIdx.y;
  if (e >= E) return;
  const int lane = threadIdx.x;
  const int s = src[e];
  const int seg = dst[e] * RR + et[e];
  const float4 v = ((const float4*)(h + (size_t)s * DD))[lane];
  float* a = agg + (size_t)seg * DD + lane * 4;
  atomicAdd(a + 0, v.x);
  atomicAdd(a + 1, v.y);
  atomicAdd(a + 2, v.z);
  atomicAdd(a + 3, v.w);
  if (lane == 0) atomicAdd(cnt + seg, 1.0f);
}

// One wave per edge: gsum[src,:] += (h[src]-h[dst])^2, gcnt[src] += 1
__global__ void ggRGCN_gate_scatter_kernel(const float* __restrict__ h,
                                           const int* __restrict__ src,
                                           const int* __restrict__ dst,
                                           float* __restrict__ gsum,
                                           float* __restrict__ gcnt, int E) {
  const int e = blockIdx.x * blockDim.y + threadIdx.y;
  if (e >= E) return;
  const int lane = threadIdx.x;
  const int s = src[e], d = dst[e];
  const float4 a = ((const float4*)(h + (size_t)s * DD))[lane];
  const float4 b = ((const float4*)(h + (size_t)d * DD))[lane];
  const float dx = a.x - b.x, dy = a.y - b.y, dz = a.z - b.z, dw = a.w - b.w;
  float* g = gsum + (size_t)s * DD + lane * 4;
  atomicAdd(g + 0, dx * dx);
  atomicAdd(g + 1, dy * dy);
  atomicAdd(g + 2, dz * dz);
  atomicAdd(g + 3, dw * dw);
  if (lane == 0) atomicAdd(gcnt + s, 1.0f);
}

// tau = tanh(gsum/max(gcnt,1)); h = (1-tau)*h + tau*hnew; also emit latent.
__global__ void ggRGCN_mix_kernel(float* __restrict__ h,
                                  const float* __restrict__ hnew,
                                  const float* __restrict__ gsum,
                                  const float* __restrict__ gcnt,
                                  float* __restrict__ lat, int total) {
  const int idx = blockIdx.x * blockDim.x + threadIdx.x;
  if (idx >= total) return;
  const float tau = tanhf(gsum[idx] / fmaxf(gcnt[idx >> 7], 1.0f));
  const float r = (1.0f - tau) * h[idx] + tau * hnew[idx];
  h[idx] = r;
  lat[idx] = r;
}

__global__ void ggRGCN_zero4_kernel(float4* __restrict__ p, long long n4) {
  const long long i = (long long)blockIdx.x * blockDim.x + threadIdx.x;
  if (i < n4) p[i] = (float4){0.0f, 0.0f, 0.0f, 0.0f};
}

__global__ void ggRGCN_copy_kernel(const float* __restrict__ a,
                                   float* __restrict__ b, long long n) {
  const long long i = (long long)blockIdx.x * blockDim.x + threadIdx.x;
  if (i < n) b[i] = a[i];
}

// ---------------------------------------------------------------------------
extern "C" void kernel_launch(void* const* d_in, const int* in_sizes, int n_in,
                              void* d_out, int out_size, void* d_ws,
                              size_t ws_size, hipStream_t stream) {
  (void)n_in; (void)out_size; (void)ws_size;

  const float* x      = (const float*)d_in[0];
  const int*   src    = (const int*)d_in[1];
  const int*   dst    = (const int*)d_in[2];
  const int*   et     = (const int*)d_in[3];
  const float* proj_W = (const float*)d_in[4];
  const float* proj_b = (const float*)d_in[5];
  const float* basisA[2] = {(const float*)d_in[6],  (const float*)d_in[10]};
  const float* compA [2] = {(const float*)d_in[7],  (const float*)d_in[11]};
  const float* rootA [2] = {(const float*)d_in[8],  (const float*)d_in[12]};
  const float* biasA [2] = {(const float*)d_in[9],  (const float*)d_in[13]};
  const float* out_W  = (const float*)d_in[14];
  const float* out_b  = (const float*)d_in[15];

  const int N = in_sizes[0] / DD;
  const int E = in_sizes[1];

  // workspace layout (all offsets 32B-aligned)
  float* ws   = (float*)d_ws;
  float* h    = ws;                                // N*DD
  float* hnew = h    + (size_t)N * DD;             // N*DD
  float* agg  = hnew + (size_t)N * DD;             // N*RR*DD
  float* cnt  = agg  + (size_t)N * RR * DD;        // N*RR
  float* gsum = cnt  + (size_t)N * RR;             // N*DD
  float* gcnt = gsum + (size_t)N * DD;             // N
  float* Wcat = gcnt + (size_t)N;                  // RR*DD*DD
  __bf16* pk_w    = (__bf16*)(Wcat + (size_t)RR * DD * DD);
  __bf16* pk_root = pk_w    + (size_t)RR * DD * DD;
  __bf16* pk_proj = pk_root + (size_t)DD * DD;
  __bf16* pk_out  = pk_proj + (size_t)DD * DD;

  // output layout: (out [N,OUTD], latents [3,N,DD]) flattened in order
  float* out  = (float*)d_out;
  float* lat0 = out  + (size_t)N * OUTD;
  float* lat1 = lat0 + (size_t)N * DD;
  float* lat2 = lat1 + (size_t)N * DD;

  const dim3 waveBlk(32, 8);                      // 8 waves / block
  const int  nTiles16 = (N + 15) / 16;            // 16-row tiles
  const int  gemmGrid = (nTiles16 + 7) / 8;
  const dim3 edgeBlk(32, 8);                      // 1 wave per edge
  const long long nd = (long long)N * DD;

  // ---- input projection: h = relu(x @ proj_W + proj_b) ----
  {
    const int pt = (DD / 32) * (DD / 16) * 32;
    ggRGCN_pack_b_kernel<<<(pt + 255) / 256, 256, 0, stream>>>(proj_W, DD, DD,
                                                               pk_proj);
    ggRGCN_dense_gemm_kernel<4, 8><<<gemmGrid, waveBlk, 0, stream>>>(
        x, pk_proj, proj_b, h, N, 1);
    ggRGCN_copy_kernel<<<(int)((nd + 255) / 256), 256, 0, stream>>>(h, lat0, nd);
  }

  // ---- two RGCN + gate layers ----
  for (int L = 0; L < 2; ++L) {
    const long long zn =
        (long long)N * RR * DD + (long long)N * RR + nd + N;  // agg..gcnt
    const long long zn4 = zn / 4;                             // zn % 4 == 0
    ggRGCN_zero4_kernel<<<(int)((zn4 + 255) / 256), 256, 0, stream>>>(
        (float4*)agg, zn4);

    ggRGCN_scatter_agg_kernel<<<(E + 7) / 8, edgeBlk, 0, stream>>>(
        h, src, dst, et, agg, cnt, E);

    ggRGCN_relw_kernel<<<(RR * DD * DD + 255) / 256, 256, 0, stream>>>(
        basisA[L], compA[L], Wcat);
    {
      const int pt1 = (RR * DD / 32) * (DD / 16) * 32;
      ggRGCN_pack_b_kernel<<<(pt1 + 255) / 256, 256, 0, stream>>>(
          Wcat, RR * DD, DD, pk_w);
      const int pt2 = (DD / 32) * (DD / 16) * 32;
      ggRGCN_pack_b_kernel<<<(pt2 + 255) / 256, 256, 0, stream>>>(
          rootA[L], DD, DD, pk_root);
    }

    // hnew = relu((agg/cnt) @ Wcat + h @ root + bias)
    ggRGCN_layer_gemm_kernel<<<gemmGrid, waveBlk, 0, stream>>>(
        agg, cnt, pk_w, h, pk_root, biasA[L], hnew, N);

    ggRGCN_gate_scatter_kernel<<<(E + 7) / 8, edgeBlk, 0, stream>>>(
        h, src, dst, gsum, gcnt, E);

    ggRGCN_mix_kernel<<<(int)((nd + 255) / 256), 256, 0, stream>>>(
        h, hnew, gsum, gcnt, (L == 0) ? lat1 : lat2, (int)nd);
  }

  // ---- readout: out = h @ out_W + out_b ----
  {
    const int pt = (DD / 32) * (OUTD / 16) * 32;
    ggRGCN_pack_b_kernel<<<(pt + 255) / 256, 256, 0, stream>>>(out_W, DD, OUTD,
                                                               pk_out);
    ggRGCN_dense_gemm_kernel<4, 4><<<gemmGrid, waveBlk, 0, stream>>>(
        h, pk_out, out_b, out, N, 0);
  }
}
